// Conjunction_Shuffle_57793079935282
// MI455X (gfx1250) — matrix operand order repaired and checked by reading, not compiled, not run on hardware
//
#include <hip/hip_runtime.h>
#include <hip/hip_bf16.h>

// ---------------------------------------------------------------------------
// Fused:  out = (x * (x >= -1)) @ W  - 0.1 * (|x| @ |W|) + 1e-5
// M=8192 (batch), K=2048 (in), N=1024 (out), all f32 in memory.
//
// Strategy: split-bf16 GEMM on V_WMMA_F32_16X16X32_BF16.
//   A-side operands (per K-tile, built on the fly from x):
//     a1 = masked x   -> a1hi + a1lo (bf16 pair)
//     a2 = |x|        -> a2hi + a2lo
//   B-side operands (built from W):
//     b1 = W          -> b1hi + b1lo
//     b2 = -0.1*|W|   -> b2hi + b2lo
//   acc += a1hi*b1hi + a1hi*b1lo + a1lo*b1hi
//        + a2hi*b2hi + a2hi*b2lo + a2lo*b2hi        (lo*lo dropped, ~2^-16 rel)
// ---------------------------------------------------------------------------

typedef __attribute__((ext_vector_type(16))) __bf16 v16bf;
typedef __attribute__((ext_vector_type(8)))  __bf16 v8bf;
typedef __attribute__((ext_vector_type(4)))  __bf16 v4bf;
typedef __attribute__((ext_vector_type(8)))  float  v8f;
typedef __attribute__((ext_vector_type(4)))  float  v4f;

static constexpr int Mdim = 8192;
static constexpr int Kdim = 2048;
static constexpr int Ndim = 1024;

static constexpr int MB = 128;   // M tile per block
static constexpr int NB = 128;   // N tile per block
static constexpr int KB = 32;    // K step (one WMMA K)

__device__ __forceinline__ void split8(const float* v, __bf16* dstH, __bf16* dstL) {
    v8bf h, l;
#pragma unroll
    for (int i = 0; i < 8; ++i) {
        __bf16 hh = (__bf16)v[i];
        h[i] = hh;
        l[i] = (__bf16)(v[i] - (float)hh);
    }
    *(v8bf*)dstH = h;
    *(v8bf*)dstL = l;
}

__global__ __launch_bounds__(256)
void fused_masked_dual_gemm(const float* __restrict__ X,
                            const float* __restrict__ W,
                            float* __restrict__ out)
{
    // LDS staging: [0]=term1 hi, [1]=term1 lo, [2]=term2 hi, [3]=term2 lo
    __shared__ __align__(32) __bf16 sA[4][MB * KB];  // row-major: [m][k], k stride 1
    __shared__ __align__(32) __bf16 sB[4][NB * KB];  // col-major: [n][k], k stride 1

    const int tid  = threadIdx.x;
    const int lane = tid & 31;
    const int wave = tid >> 5;        // 0..7
    const int wm   = wave >> 2;       // 0..1  -> 64-row half
    const int wn   = wave & 3;        // 0..3  -> 32-col quarter

    const int n_tile = blockIdx.x * NB;
    const int m_tile = blockIdx.y * MB;

    // --- staging decomposition ---
    // A: 128 rows x 32 k = 4096 f32; 2 threads/row, 16 f32 each.
    const int a_row = tid >> 1;           // 0..127
    const int a_k0  = (tid & 1) * 16;     // 0 or 16
    // B: 32 k x 128 n = 4096 f32; 4x4 micro-blocks, 256 threads.
    const int b_k0  = (tid >> 5) * 4;     // 0..28
    const int b_n0  = (tid & 31) * 4;     // 0..124

    v8f acc[4][2];
#pragma unroll
    for (int mf = 0; mf < 4; ++mf)
#pragma unroll
        for (int nf = 0; nf < 2; ++nf)
            acc[mf][nf] = v8f{0.f,0.f,0.f,0.f,0.f,0.f,0.f,0.f};

    // fragment lane geometry (wave32; A and B both use contiguous k = 16*(lane/16) + s)
    const int lrow = lane & 15;
    const int lka  = (lane >> 4) * 16;    // A: chunk0 at K=lka, chunk1 at K=lka+8
    const int lkb  = (lane >> 4) * 16;    // B: 16 contiguous K per half-wave

    for (int kb = 0; kb < Kdim; kb += KB) {
        // ---------------- stage A tile (x -> masked / abs, hi/lo bf16) ------
        {
            const float* xp = X + (size_t)(m_tile + a_row) * Kdim + kb + a_k0;
            float xv[16];
#pragma unroll
            for (int j = 0; j < 4; ++j) {
                v4f t = *reinterpret_cast<const v4f*>(xp + 4 * j);
                xv[4*j+0] = t.x; xv[4*j+1] = t.y; xv[4*j+2] = t.z; xv[4*j+3] = t.w;
            }
            float a1[16], a2[16];
#pragma unroll
            for (int i = 0; i < 16; ++i) {
                float v = xv[i];
                a1[i] = (v >= -1.0f) ? v : 0.0f;   // mask applied pre-GEMM
                a2[i] = __builtin_fabsf(v);
            }
            const int base = a_row * KB + a_k0;
            split8(a1 + 0, &sA[0][base + 0], &sA[1][base + 0]);
            split8(a1 + 8, &sA[0][base + 8], &sA[1][base + 8]);
            split8(a2 + 0, &sA[2][base + 0], &sA[3][base + 0]);
            split8(a2 + 8, &sA[2][base + 8], &sA[3][base + 8]);
            if (kb + KB < Kdim)
                __builtin_prefetch(xp + KB, 0, 3);   // global_prefetch_b8
        }
        // ---------------- stage B tile (W -> W / -0.1|W|, transposed) -------
        {
            float wblk[4][4];
#pragma unroll
            for (int r = 0; r < 4; ++r) {
                const float* wp = W + (size_t)(kb + b_k0 + r) * Ndim + n_tile + b_n0;
                v4f t = *reinterpret_cast<const v4f*>(wp);
                wblk[r][0] = t.x; wblk[r][1] = t.y; wblk[r][2] = t.z; wblk[r][3] = t.w;
                if (kb + KB < Kdim)
                    __builtin_prefetch(wp + (size_t)KB * Ndim, 0, 3);
            }
#pragma unroll
            for (int c = 0; c < 4; ++c) {
                v4bf h1, l1, h2, l2;
#pragma unroll
                for (int r = 0; r < 4; ++r) {
                    float b1 = wblk[r][c];
                    float b2 = -0.1f * __builtin_fabsf(b1);
                    __bf16 hh1 = (__bf16)b1;
                    __bf16 hh2 = (__bf16)b2;
                    h1[r] = hh1; l1[r] = (__bf16)(b1 - (float)hh1);
                    h2[r] = hh2; l2[r] = (__bf16)(b2 - (float)hh2);
                }
                const int base = (b_n0 + c) * KB + b_k0;   // [n][k], 8B aligned
                *(v4bf*)&sB[0][base] = h1;
                *(v4bf*)&sB[1][base] = l1;
                *(v4bf*)&sB[2][base] = h2;
                *(v4bf*)&sB[3][base] = l2;
            }
        }
        __syncthreads();

        // ---------------- compute: 48 WMMA per wave per K-step --------------
        v16bf bf[4][2];
#pragma unroll
        for (int nf = 0; nf < 2; ++nf) {
            const int col = wn * 32 + nf * 16 + lrow;
#pragma unroll
            for (int t = 0; t < 4; ++t)
                bf[t][nf] = *(const v16bf*)&sB[t][col * KB + lkb];
        }

#pragma unroll
        for (int mf = 0; mf < 4; ++mf) {
            const int row = wm * 64 + mf * 16 + lrow;
            v16bf af[4];
#pragma unroll
            for (int t = 0; t < 4; ++t) {
                v8bf c0 = *(const v8bf*)&sA[t][row * KB + lka];
                v8bf c1 = *(const v8bf*)&sA[t][row * KB + lka + 8];
                af[t] = __builtin_shufflevector(c0, c1,
                          0,1,2,3,4,5,6,7,8,9,10,11,12,13,14,15);
            }
#pragma unroll
            for (int nf = 0; nf < 2; ++nf) {
                v8f c = acc[mf][nf];
                // term 1: masked-x @ W   (hi*hi + hi*lo + lo*hi)
                c = __builtin_amdgcn_wmma_f32_16x16x32_bf16(false, af[0], false, bf[0][nf], (short)0, c, false, false);
                c = __builtin_amdgcn_wmma_f32_16x16x32_bf16(false, af[0], false, bf[1][nf], (short)0, c, false, false);
                c = __builtin_amdgcn_wmma_f32_16x16x32_bf16(false, af[1], false, bf[0][nf], (short)0, c, false, false);
                // term 2: |x| @ (-0.1*|W|)
                c = __builtin_amdgcn_wmma_f32_16x16x32_bf16(false, af[2], false, bf[2][nf], (short)0, c, false, false);
                c = __builtin_amdgcn_wmma_f32_16x16x32_bf16(false, af[2], false, bf[3][nf], (short)0, c, false, false);
                c = __builtin_amdgcn_wmma_f32_16x16x32_bf16(false, af[3], false, bf[2][nf], (short)0, c, false, false);
                acc[mf][nf] = c;
            }
        }
        __syncthreads();
    }

    // ---------------- epilogue: + DELTA*MAX_ABS_W, store C ------------------
    const float cbias = 0.1f * 1e-4f;    // 1e-5
#pragma unroll
    for (int mf = 0; mf < 4; ++mf) {
#pragma unroll
        for (int nf = 0; nf < 2; ++nf) {
            const int col = n_tile + wn * 32 + nf * 16 + (lane & 15);
            const int rbase = m_tile + wm * 64 + mf * 16 + (lane >> 4) * 8;
#pragma unroll
            for (int r = 0; r < 8; ++r)
                out[(size_t)(rbase + r) * Ndim + col] = acc[mf][nf][r] + cbias;
        }
    }
}

extern "C" void kernel_launch(void* const* d_in, const int* in_sizes, int n_in,
                              void* d_out, int out_size, void* d_ws, size_t ws_size,
                              hipStream_t stream) {
    (void)in_sizes; (void)n_in; (void)out_size; (void)d_ws; (void)ws_size;
    const float* x = (const float*)d_in[0];   // (8192, 2048) f32
    const float* W = (const float*)d_in[1];   // (2048, 1024) f32
    float* out = (float*)d_out;               // (8192, 1024) f32

    dim3 grid(Ndim / NB, Mdim / MB);          // (8, 64)
    dim3 block(256);                          // 8 wave32 waves
    fused_masked_dual_gemm<<<grid, block, 0, stream>>>(x, W, out);
}
